// Mamba2DScalarHead_86904368267767
// MI455X (gfx1250) — matrix operand order repaired and checked by reading, not compiled, not run on hardware
//
#include <hip/hip_runtime.h>
#include <hip/hip_bf16.h>

// ---------------------------------------------------------------------------
// Mamba-2D scalar head for MI455X (gfx1250, wave32).
// GEMMs: v_wmma_f32_16x16x32_bf16, 32x64 register tile per wave.
// All GEMM operands (activations + weights) are pre-converted to bf16 at the
// producer, so the GEMM K-loop is pure b128 loads + WMMA.
// ---------------------------------------------------------------------------

#define BB    16          // batch
#define HP    32
#define WP    32
#define LL    (HP * WP)   // 1024 tokens
#define CC    768
#define DD    512
#define MROWS (BB * LL)   // 16384 rows

typedef __attribute__((ext_vector_type(16))) __bf16 v16bf;
typedef __attribute__((ext_vector_type(8)))  float  v8f;

union BF16x16 {
    v16bf          v;
    unsigned short s[16];
    uint4          q[2];
};

__device__ __forceinline__ unsigned short f2bf(float f) {
    // round-to-nearest-even fp32 -> bf16
    unsigned int u = __float_as_uint(f);
    unsigned int r = u + 0x7FFFu + ((u >> 16) & 1u);
    return (unsigned short)(r >> 16);
}

__device__ __forceinline__ float bf2f(unsigned short s) {
    return __uint_as_float((unsigned int)s << 16);
}

__device__ __forceinline__ float sigmoidf(float x) {
    return 1.0f / (1.0f + __expf(-x));
}

// ---------------------------------------------------------------------------
// fp32 -> bf16 elementwise convert (used once for tokens).
// ---------------------------------------------------------------------------
__global__ void f32_to_bf16_kernel(const float* __restrict__ x,
                                   unsigned short* __restrict__ o, int n)
{
    int idx = blockIdx.x * blockDim.x + threadIdx.x;
    if (idx >= n) return;
    o[idx] = f2bf(x[idx]);
}

// ---------------------------------------------------------------------------
// Pack a row-major fp32 weight matrix W[K][N] into bf16 WMMA B-fragments.
// Layout: [kt][nt][lane][16 bf16]; lanes 0-15 hold K rows kt*32+{0..7,16..23},
// lanes 16-31 hold K rows kt*32+{8..15,24..31}, column = nt*16 + lane%16.
// ---------------------------------------------------------------------------
__global__ void pack_w_kernel(const float* __restrict__ W,
                              unsigned short* __restrict__ P,
                              int K, int N)
{
    int tid   = blockIdx.x * blockDim.x + threadIdx.x;
    int total = (K >> 5) * (N >> 4) * 32;
    if (tid >= total) return;
    int lane   = tid & 31;
    int t      = tid >> 5;
    int ntiles = N >> 4;
    int nt     = t % ntiles;
    int kt     = t / ntiles;
    int col  = (nt << 4) + (lane & 15);
    int krow = (kt << 5) + ((lane < 16) ? 0 : 8);
    unsigned short* o = P + ((size_t)tid << 4);
#pragma unroll
    for (int i = 0; i < 16; ++i)
        o[i] = f2bf(W[(size_t)(krow + (i < 8 ? i : i + 8)) * N + col]);
}

// ---------------------------------------------------------------------------
// C[M][N] = A[M][K](bf16 row-major) @ Bp + bias (+= C when ACC).
// Each wave computes a 32x64 tile: 2 M-subtiles x 4 N-subtiles, K step 32.
// Inner loop: 12x global_load_b128 + 8x v_wmma_f32_16x16x32_bf16.
// ---------------------------------------------------------------------------
template <bool ACC>
__global__ void gemm_bf16_wmma(const unsigned short* __restrict__ A,
                               const unsigned short* __restrict__ Bp,
                               const float* __restrict__ bias,
                               float* __restrict__ C,
                               int M, int N, int K)
{
    const int lane    = threadIdx.x & 31;
    const int wid     = blockIdx.x * (blockDim.x >> 5) + (threadIdx.x >> 5);
    const int ntiles  = N >> 4;       // 16-col tiles
    const int ngroups = ntiles >> 2;  // 4 tiles per wave
    const int mgroups = M >> 5;       // 2 row-tiles per wave
    if (wid >= mgroups * ngroups) return;
    const int mg = wid / ngroups;
    const int ng = wid % ngroups;

    const int l15   = lane & 15;
    const int hi    = lane >> 4;                 // lane half
    const int koff  = hi ? 8 : 0;                // A-frag K offset per half
    const int arow0 = (mg << 5) + l15;
    const int nt0   = ng << 2;

    v8f acc[2][4] = {};

    for (int k = 0; k < K; k += 32) {
        // ---- two A fragments (rows arow0, arow0+16): 2 b128 loads each ----
        BF16x16 a[2];
#pragma unroll
        for (int mi = 0; mi < 2; ++mi) {
            const unsigned short* ap =
                A + (size_t)(arow0 + mi * 16) * K + k + koff;
            a[mi].q[0] = *(const uint4*)(ap);        // K = k+koff   .. +7
            a[mi].q[1] = *(const uint4*)(ap + 16);   // K = k+koff+16.. +23
        }
        // ---- four B fragments, each reused across both M subtiles ----
#pragma unroll
        for (int j = 0; j < 4; ++j) {
            BF16x16 b;
            const uint4* bp = (const uint4*)(Bp +
                ((size_t)(((k >> 5) * ntiles + nt0 + j) * 32 + lane) << 4));
            b.q[0] = bp[0];
            b.q[1] = bp[1];
            acc[0][j] = __builtin_amdgcn_wmma_f32_16x16x32_bf16(
                false, a[0].v, false, b.v, (short)0, acc[0][j], false, false);
            acc[1][j] = __builtin_amdgcn_wmma_f32_16x16x32_bf16(
                false, a[1].v, false, b.v, (short)0, acc[1][j], false, false);
        }
    }

    // ---- branchless epilogue (C/D layout: VGPR v -> row rbase+v, col l15) ----
    const int rbase = (mg << 5) + (hi ? 8 : 0);
#pragma unroll
    for (int mi = 0; mi < 2; ++mi) {
#pragma unroll
        for (int j = 0; j < 4; ++j) {
            const int col = ((nt0 + j) << 4) + l15;
            const float bv = bias[col];
            float* cp = C + (size_t)(rbase + mi * 16) * N + col;
#pragma unroll
            for (int v = 0; v < 8; ++v) {
                size_t o   = (size_t)v * N;
                float  val = acc[mi][j][v] + bv;
                if (ACC) val += cp[o];
                cp[o] = val;
            }
        }
    }
}

// ---------------------------------------------------------------------------
// Row LayerNorm over D=512; one wave per row (wave32 shfl reduction).
// Output stored as bf16 (feeds WMMA GEMMs / pooling).
// ---------------------------------------------------------------------------
__global__ void ln_rows_kernel(const float* __restrict__ x,
                               const float* __restrict__ w,
                               const float* __restrict__ b,
                               unsigned short* __restrict__ out, int rows)
{
    int lane = threadIdx.x & 31;
    int row  = blockIdx.x * (blockDim.x >> 5) + (threadIdx.x >> 5);
    if (row >= rows) return;
    const float* xr = x + (size_t)row * DD;
    float vals[16];
    float s = 0.0f, sq = 0.0f;
#pragma unroll
    for (int i = 0; i < 16; ++i) {
        float v = xr[lane + 32 * i];
        vals[i] = v;
        s  += v;
        sq += v * v;
    }
#pragma unroll
    for (int off = 16; off >= 1; off >>= 1) {
        s  += __shfl_xor(s,  off, 32);
        sq += __shfl_xor(sq, off, 32);
    }
    float mu  = s * (1.0f / DD);
    float var = sq * (1.0f / DD) - mu * mu;
    float inv = rsqrtf(var + 1e-5f);
    unsigned short* orow = out + (size_t)row * DD;
#pragma unroll
    for (int i = 0; i < 16; ++i) {
        int c = lane + 32 * i;
        orow[c] = f2bf((vals[i] - mu) * inv * w[c] + b[c]);
    }
}

// ---------------------------------------------------------------------------
// Depthwise 3x3 SAME conv + bias + SiLU. Input = first D cols of xg (stride 2D).
// ---------------------------------------------------------------------------
__global__ void dwconv_silu_kernel(const float* __restrict__ xg,
                                   const float* __restrict__ kw,  // (3,3,1,D)
                                   const float* __restrict__ kb,
                                   float* __restrict__ u)
{
    int idx = blockIdx.x * blockDim.x + threadIdx.x;
    if (idx >= MROWS * DD) return;
    int d = idx & (DD - 1);
    int w = (idx >> 9)  & (WP - 1);
    int h = (idx >> 14) & (HP - 1);
    int b = idx >> 19;
    float acc = 0.0f;
#pragma unroll
    for (int kh = 0; kh < 3; ++kh) {
        int ih = h + kh - 1;
        if (ih < 0 || ih >= HP) continue;
#pragma unroll
        for (int kwi = 0; kwi < 3; ++kwi) {
            int iw = w + kwi - 1;
            if (iw < 0 || iw >= WP) continue;
            size_t xo = ((size_t)(((b * HP + ih) * WP + iw)) << 10) + d; // row * 2D
            acc += xg[xo] * kw[(kh * 3 + kwi) * DD + d];
        }
    }
    float v = acc + kb[d];
    u[idx] = v * sigmoidf(v);   // SiLU
}

// ---------------------------------------------------------------------------
// Bidirectional row+col SSM scan, averaged (x0.25). One thread per (b,d)
// runs all 4 directions sequentially -> deterministic, no atomics.
// s_t = a*s_{t-1} + b*x_t ; y_t += 0.25*(c*s_t + d*x_t)
// ---------------------------------------------------------------------------
__global__ void scan2d_kernel(const float* __restrict__ u,
                              const float* __restrict__ al,
                              const float* __restrict__ bbp,
                              const float* __restrict__ ccp,
                              const float* __restrict__ ddp,
                              float* __restrict__ y)
{
    int tid = blockIdx.x * blockDim.x + threadIdx.x;
    if (tid >= BB * DD) return;
    int d = tid & (DD - 1);
    int b = tid >> 9;

    float a = sigmoidf(al[d]);
    a = fminf(fmaxf(a, 1e-4f), 1.0f - 1e-4f);
    const float bv = bbp[d], cv = ccp[d], dv = ddp[d];
    const size_t base = ((size_t)b << 10) * DD + d;

    for (int dir = 0; dir < 4; ++dir) {
        float s = 0.0f;
        for (int t = 0; t < LL; ++t) {
            int tt = (dir & 1) ? (LL - 1 - t) : t;
            int pos;
            if (dir < 2) {
                pos = tt;                                  // row-major
            } else {
                int wq = tt >> 5, hq = tt & 31;            // col-major order
                pos = hq * WP + wq;
            }
            size_t o = base + (size_t)pos * DD;
            float  x = u[o];
            s = a * s + bv * x;
            float val = 0.25f * (cv * s + dv * x);
            if (dir == 0) y[o] = val;
            else          y[o] += val;
        }
    }
}

// ---------------------------------------------------------------------------
// yg = bf16(y * sigmoid(gate)); gate = second D cols of xg.
// ---------------------------------------------------------------------------
__global__ void gate_mul_kernel(const float* __restrict__ y,
                                const float* __restrict__ xg,
                                unsigned short* __restrict__ yg)
{
    int idx = blockIdx.x * blockDim.x + threadIdx.x;
    if (idx >= MROWS * DD) return;
    int row = idx >> 9;
    int c   = idx & (DD - 1);
    float g = sigmoidf(xg[((size_t)row << 10) + DD + c]);
    yg[idx] = f2bf(y[idx] * g);
}

// ---------------------------------------------------------------------------
// Mean pool over the N=1024 tokens of each batch: out[b][d].
// ---------------------------------------------------------------------------
__global__ void pool_kernel(const unsigned short* __restrict__ h,
                            float* __restrict__ out)
{
    int idx = blockIdx.x * blockDim.x + threadIdx.x;
    if (idx >= BB * DD) return;
    int d = idx & (DD - 1);
    int b = idx >> 9;
    float s = 0.0f;
    for (int n = 0; n < LL; ++n)
        s += bf2f(h[(((size_t)b << 10) + n) * DD + d]);
    out[idx] = s * (1.0f / LL);
}

// ---------------------------------------------------------------------------
extern "C" void kernel_launch(void* const* d_in, const int* in_sizes, int n_in,
                              void* d_out, int out_size, void* d_ws, size_t ws_size,
                              hipStream_t stream)
{
    (void)in_sizes; (void)n_in; (void)out_size; (void)ws_size;

    const float* tokens = (const float*)d_in[0];
    const float* in_w   = (const float*)d_in[1];
    const float* in_b   = (const float*)d_in[2];
    const float* nw     = (const float*)d_in[3];
    const float* nb     = (const float*)d_in[4];
    const float* iw     = (const float*)d_in[5];
    const float* ib     = (const float*)d_in[6];
    const float* cw     = (const float*)d_in[7];
    const float* cb     = (const float*)d_in[8];
    const float* al     = (const float*)d_in[9];
    const float* sb     = (const float*)d_in[10];
    const float* sc     = (const float*)d_in[11];
    const float* sd     = (const float*)d_in[12];
    const float* ow     = (const float*)d_in[13];
    const float* ob     = (const float*)d_in[14];
    const float* onw    = (const float*)d_in[15];
    const float* onb    = (const float*)d_in[16];
    float* out = (float*)d_out;

    // ---- workspace layout (bytes) ----
    char* ws = (char*)d_ws;
    const size_t SZ_PK_IN  = (size_t)CC * DD * 2;            //  768 KB
    const size_t SZ_PK_INP = (size_t)4 * DD * (2 * DD) * 2;  //  4 MB
    const size_t SZ_PK_OUT = (size_t)4 * DD * DD * 2;        //  2 MB
    const size_t SZ_TOK_BF = (size_t)MROWS * CC * 2;         // 24 MB
    const size_t SZ_H_BF   = (size_t)MROWS * DD * 2;         // 16 MB
    const size_t SZ_ACT    = (size_t)MROWS * DD * 4;         // 32 MB
    const size_t SZ_XG     = (size_t)MROWS * (2 * DD) * 4;   // 64 MB

    unsigned short* pk_in  = (unsigned short*)(ws);
    unsigned short* pk_inp = (unsigned short*)(ws + SZ_PK_IN);
    unsigned short* pk_out = (unsigned short*)(ws + SZ_PK_IN + SZ_PK_INP);
    size_t fo = SZ_PK_IN + SZ_PK_INP + SZ_PK_OUT;
    unsigned short* tok_bf = (unsigned short*)(ws + fo);  fo += SZ_TOK_BF;
    unsigned short* hbuf   = (unsigned short*)(ws + fo);  fo += SZ_H_BF;  // LN / yg
    float* xbuf = (float*)(ws + fo);                      fo += SZ_ACT;
    float* xg   = (float*)(ws + fo);                      fo += SZ_XG;
    float* ubuf = (float*)(ws + fo);                      fo += SZ_ACT;
    float* ybuf = (float*)(ws + fo);

    // ---- one-time conversions/packing ----
    f32_to_bf16_kernel<<<MROWS * CC / 256, 256, 0, stream>>>(
        tokens, tok_bf, MROWS * CC);
    pack_w_kernel<<<(CC / 32) * (DD / 16) * 32 / 256, 256, 0, stream>>>(
        in_w, pk_in, CC, DD);
    for (int i = 0; i < 4; ++i) {
        pack_w_kernel<<<(DD / 32) * (2 * DD / 16) * 32 / 256, 256, 0, stream>>>(
            iw + (size_t)i * DD * (2 * DD), pk_inp + (size_t)i * DD * (2 * DD),
            DD, 2 * DD);
        pack_w_kernel<<<(DD / 32) * (DD / 16) * 32 / 256, 256, 0, stream>>>(
            ow + (size_t)i * DD * DD, pk_out + (size_t)i * DD * DD, DD, DD);
    }

    const int EW_BLOCKS = MROWS * DD / 256;                  // 32768
    const int LN_BLOCKS = MROWS / 8;                         // 2048

    // waves per GEMM: (M/32) * (N/64); 8 waves per block
    const int GEMM_D_BLOCKS  = (MROWS / 32) * (DD / 64) / 8;       // 512
    const int GEMM_2D_BLOCKS = (MROWS / 32) * (2 * DD / 64) / 8;   // 1024

    // ---- input projection: xbuf = tokens @ in_w + in_b ----
    gemm_bf16_wmma<false><<<GEMM_D_BLOCKS, 256, 0, stream>>>(
        tok_bf, pk_in, in_b, xbuf, MROWS, DD, CC);

    // ---- 4 Mamba-2D blocks ----
    for (int i = 0; i < 4; ++i) {
        ln_rows_kernel<<<LN_BLOCKS, 256, 0, stream>>>(
            xbuf, nw + i * DD, nb + i * DD, hbuf, MROWS);

        gemm_bf16_wmma<false><<<GEMM_2D_BLOCKS, 256, 0, stream>>>(
            hbuf, pk_inp + (size_t)i * DD * (2 * DD), ib + i * (2 * DD),
            xg, MROWS, 2 * DD, DD);

        dwconv_silu_kernel<<<EW_BLOCKS, 256, 0, stream>>>(
            xg, cw + (size_t)i * 9 * DD, cb + i * DD, ubuf);

        scan2d_kernel<<<BB * DD / 256, 256, 0, stream>>>(
            ubuf, al + i * DD, sb + i * DD, sc + i * DD, sd + i * DD, ybuf);

        gate_mul_kernel<<<EW_BLOCKS, 256, 0, stream>>>(ybuf, xg, hbuf);

        gemm_bf16_wmma<true><<<GEMM_D_BLOCKS, 256, 0, stream>>>(
            hbuf, pk_out + (size_t)i * DD * DD, ob + i * DD,
            xbuf, MROWS, DD, DD);   // residual accumulate
    }

    // ---- final LayerNorm + mean pool ----
    ln_rows_kernel<<<LN_BLOCKS, 256, 0, stream>>>(xbuf, onw, onb, hbuf, MROWS);
    pool_kernel<<<BB * DD / 256, 256, 0, stream>>>(hbuf, out);
}
